// rnn_model_89283780149388
// MI455X (gfx1250) — compile-verified, matrix operand order checked by reading, not run on hardware
//
#include <hip/hip_runtime.h>
#include <stdint.h>

// Problem constants (from reference): T=256, B=16, H=512, V=32000
#define T_LEN 256
#define BATCH 16
#define HDIM  512
#define VOCAB 32000
#define NSAMP (T_LEN * BATCH)   // 4096 rows of decode output

typedef __attribute__((ext_vector_type(16))) __bf16 v16bf;
typedef __attribute__((ext_vector_type(8)))  float  v8f;

union FragU { uint4  u[2]; v16bf v; };   // 32B: one bf16 16x32 A/B fragment per lane
union AccU  { float4 f[2]; v8f   v; };   // 32B: one f32 16x16 C/D fragment per lane

// round-to-nearest-even f32 -> bf16 bits
__device__ __forceinline__ unsigned short f2bf(float x) {
    unsigned u = __float_as_uint(x);
    unsigned r = u + 0x7FFFu + ((u >> 16) & 1u);
    return (unsigned short)(r >> 16);
}

// A/B fragment loader for 16x32 bf16 tiles (ISA 7.12.2 layout):
// lane holds row (lane&15); dwords 0..3 = K[k0+hi*8 .. +7], dwords 4..7 = K[k0+16+hi*8 .. +7]
__device__ __forceinline__ v16bf load_frag(const unsigned short* p) {
    FragU f;
    f.u[0] = *(const uint4*)(p);
    f.u[1] = *(const uint4*)(p + 16);   // +16 bf16 elements = +32B in K
    return f.v;
}

// ---------------------------------------------------------------------------
// Kernel 1: convert dec_w / w1 to bf16, gather+convert embeddings
// ---------------------------------------------------------------------------
__global__ void prep_kernel(const float* __restrict__ dec_w,
                            const float* __restrict__ w1,
                            const float* __restrict__ emb,
                            const int*   __restrict__ inp,
                            unsigned short* __restrict__ dec_w_bf,
                            unsigned short* __restrict__ wx_bf,
                            unsigned short* __restrict__ wh_bf,
                            unsigned short* __restrict__ x_bf) {
    const int stride = gridDim.x * blockDim.x;
    const int tid0 = blockIdx.x * blockDim.x + threadIdx.x;

    for (int i = tid0; i < VOCAB * HDIM; i += stride)
        dec_w_bf[i] = f2bf(dec_w[i]);

    for (int i = tid0; i < HDIM * HDIM; i += stride) {
        int j = i >> 9, k = i & (HDIM - 1);
        wx_bf[i] = f2bf(w1[j * (2 * HDIM) + k]);          // wx[j][k] = w1[j][k]
        wh_bf[i] = f2bf(w1[j * (2 * HDIM) + HDIM + k]);   // wh[j][k] = w1[j][H+k]
    }

    for (int i = tid0; i < NSAMP * HDIM; i += stride) {
        int n = i >> 9, k = i & (HDIM - 1);
        x_bf[i] = f2bf(emb[(size_t)inp[n] * HDIM + k]);
    }
}

// ---------------------------------------------------------------------------
// Kernel 2: pre[n][j] = sum_k x[n,k]*wx[j,k] + b1[j]   (parallel, 2.1 GFLOP)
// one wave per 16x16 output tile; A = wx rows, B = x rows (K-contiguous)
// ---------------------------------------------------------------------------
__global__ void __launch_bounds__(256) pre_gemm_kernel(
        const unsigned short* __restrict__ wx_bf,
        const unsigned short* __restrict__ x_bf,
        const float* __restrict__ b1,
        float* __restrict__ pre) {
    const int lane = threadIdx.x & 31, wave = threadIdx.x >> 5;
    const int n16 = lane & 15, hi = lane >> 4;
    const int gw = blockIdx.x * 8 + wave;   // 0..8191
    const int jt = gw & 31;                 // 32 hidden tiles
    const int st = gw >> 5;                 // 256 sample tiles

    const unsigned short* ap = wx_bf + (jt * 16 + n16) * HDIM + hi * 8;
    const unsigned short* bp = x_bf  + (st * 16 + n16) * HDIM + hi * 8;

    v8f c = {0.f, 0.f, 0.f, 0.f, 0.f, 0.f, 0.f, 0.f};
#pragma unroll
    for (int ch = 0; ch < 16; ++ch) {
        v16bf a = load_frag(ap + ch * 32);
        v16bf b = load_frag(bp + ch * 32);
        c = __builtin_amdgcn_wmma_f32_16x16x32_bf16(false, a, false, b,
                                                    (short)0, c, false, false);
    }
    const int jbase = jt * 16 + hi * 8;
    const int samp  = st * 16 + n16;
    float4 bb0 = *(const float4*)(b1 + jbase);
    float4 bb1 = *(const float4*)(b1 + jbase + 4);
    AccU A; A.v = c;
    A.f[0].x += bb0.x; A.f[0].y += bb0.y; A.f[0].z += bb0.z; A.f[0].w += bb0.w;
    A.f[1].x += bb1.x; A.f[1].y += bb1.y; A.f[1].z += bb1.z; A.f[1].w += bb1.w;
    *(float4*)(pre + (size_t)samp * HDIM + jbase)     = A.f[0];
    *(float4*)(pre + (size_t)samp * HDIM + jbase + 4) = A.f[1];
}

// ---------------------------------------------------------------------------
// Kernel 3: serial recurrence h = tanh(pre[t] + Wh . h), single 32-wave WG.
// Each wave owns 16 output rows; its Wh tile (16x512 bf16) stays in 128 VGPRs.
// h (16x512 bf16) double-buffered in LDS; one barrier per step.
// ---------------------------------------------------------------------------
__global__ void __launch_bounds__(1024) rnn_kernel(
        const unsigned short* __restrict__ wh_bf,
        const float* __restrict__ pre,
        unsigned short* __restrict__ out_bf,
        float* __restrict__ h_last) {
    __shared__ __attribute__((aligned(16))) unsigned short hbuf[2][16][520]; // +8 pad/row

    const int tid  = threadIdx.x;
    const int lane = tid & 31, wave = tid >> 5;
    const int n16 = lane & 15, hi = lane >> 4;
    const int jbase = wave * 16 + hi * 8;

    for (int i = tid; i < 2 * 16 * 520; i += 1024)
        ((unsigned short*)hbuf)[i] = 0;

    // VGPR-resident Wh fragments: 16 chunks x 8 VGPRs = 128 VGPRs/lane
    v16bf wf[16];
    const unsigned short* wp = wh_bf + (wave * 16 + n16) * HDIM + hi * 8;
#pragma unroll
    for (int ch = 0; ch < 16; ++ch) wf[ch] = load_frag(wp + ch * 32);

    __syncthreads();

    int cur = 0;
#pragma unroll 1
    for (int t = 0; t < T_LEN; ++t) {
        const int samp = t * 16 + n16;
        AccU acc;
        acc.f[0] = *(const float4*)(pre + (size_t)samp * HDIM + jbase);
        acc.f[1] = *(const float4*)(pre + (size_t)samp * HDIM + jbase + 4);
        v8f c = acc.v;

        const unsigned short* hp = &hbuf[cur][n16][hi * 8];
#pragma unroll
        for (int ch = 0; ch < 16; ++ch) {
            v16bf b = load_frag(hp + ch * 32);
            c = __builtin_amdgcn_wmma_f32_16x16x32_bf16(false, wf[ch], false, b,
                                                        (short)0, c, false, false);
        }

        float hv[8];
#pragma unroll
        for (int g = 0; g < 8; ++g) {           // tanh(x) = 1 - 2/(e^{2x}+1)
            float e = __expf(2.0f * c[g]);
            hv[g] = 1.0f - 2.0f / (e + 1.0f);
        }
        uint4 pk;
        pk.x = f2bf(hv[0]) | ((unsigned)f2bf(hv[1]) << 16);
        pk.y = f2bf(hv[2]) | ((unsigned)f2bf(hv[3]) << 16);
        pk.z = f2bf(hv[4]) | ((unsigned)f2bf(hv[5]) << 16);
        pk.w = f2bf(hv[6]) | ((unsigned)f2bf(hv[7]) << 16);

        const int nxt = cur ^ 1;
        *(uint4*)(&hbuf[nxt][n16][jbase]) = pk;                      // next h (LDS)
        *(uint4*)(out_bf + (size_t)samp * HDIM + jbase) = pk;        // decoder input

        if (t == T_LEN - 1) {
            float4 o0 = {hv[0], hv[1], hv[2], hv[3]};
            float4 o1 = {hv[4], hv[5], hv[6], hv[7]};
            *(float4*)(h_last + n16 * HDIM + jbase)     = o0;
            *(float4*)(h_last + n16 * HDIM + jbase + 4) = o1;
        }
        __syncthreads();
        cur = nxt;
    }
}

// ---------------------------------------------------------------------------
// Kernel 4: decoder GEMM logits[n][v] = out[n,:].dec_w[v,:] + dec_b[v]
// WG = 16 waves -> 256 vocab x 128 samples. The B block (128x512 bf16, 130KB)
// is staged into LDS with GLOBAL_LOAD_ASYNC_TO_LDS_B128 (ASYNCcnt path), rows
// padded to 1040B so wave-wide ds_load_b128 hits the 2-cycle bank floor.
// A (dec_w) streams from L2; wave = 4 vtiles x 2 stiles (8 WMMA per k-chunk).
// ---------------------------------------------------------------------------
__global__ void __launch_bounds__(512) decoder_kernel(
        const unsigned short* __restrict__ dec_w_bf,
        const unsigned short* __restrict__ out_bf,
        const float* __restrict__ dec_b,
        float* __restrict__ logits) {
    __shared__ __attribute__((aligned(16))) unsigned short bstage[128 * 520]; // 133,120 B

    const int tid  = threadIdx.x;
    const int lane = tid & 31, wave = tid >> 5;
    const int n16 = lane & 15, hi = lane >> 4;
    const int wv = wave & 3;      // 4 wave-columns over vocab (4 vtiles each)
    const int ws = wave >> 2;     // 4 wave-rows over samples (2 stiles each)
    const int vt0   = blockIdx.x * 16 + wv * 4;
    const int sbase = blockIdx.y * 128;

    // --- async-stage B block: 128 rows x 64 x 16B segments -> LDS ---
    for (int s = tid; s < 128 * 64; s += 512) {
        const int r = s >> 6, seg = s & 63;
        const unsigned short* src = out_bf + (size_t)(sbase + r) * HDIM + seg * 8;
        unsigned dst = (unsigned)(size_t)(&bstage[r * 520 + seg * 8]);
        asm volatile("global_load_async_to_lds_b128 %0, %1, off"
                     :: "v"(dst), "v"(src) : "memory");
    }
    asm volatile("s_wait_asynccnt 0x0" ::: "memory");
    __syncthreads();

    const unsigned short* ap = dec_w_bf + ((size_t)(vt0 * 16 + n16)) * HDIM + hi * 8;
    const unsigned short* bp = &bstage[(ws * 2 * 16 + n16) * 520 + hi * 8];

    const v8f z = {0.f, 0.f, 0.f, 0.f, 0.f, 0.f, 0.f, 0.f};
    v8f acc[4][2];
#pragma unroll
    for (int i = 0; i < 4; ++i)
#pragma unroll
        for (int j = 0; j < 2; ++j) acc[i][j] = z;

#pragma unroll 4
    for (int ch = 0; ch < 16; ++ch) {
        v16bf a[4], b[2];
#pragma unroll
        for (int i = 0; i < 4; ++i) {
            a[i] = load_frag(ap + (size_t)i * 16 * HDIM + ch * 32);
            __builtin_prefetch(ap + (size_t)i * 16 * HDIM + ch * 32 + 32, 0, 3);
        }
#pragma unroll
        for (int j = 0; j < 2; ++j) {
            b[j] = load_frag(bp + (size_t)j * 16 * 520 + ch * 32);
        }
#pragma unroll
        for (int i = 0; i < 4; ++i)
#pragma unroll
            for (int j = 0; j < 2; ++j)
                acc[i][j] = __builtin_amdgcn_wmma_f32_16x16x32_bf16(
                    false, a[i], false, b[j], (short)0, acc[i][j], false, false);
    }

#pragma unroll
    for (int i = 0; i < 4; ++i) {
        const int vb = (vt0 + i) * 16 + hi * 8;
        float4 d0 = *(const float4*)(dec_b + vb);
        float4 d1 = *(const float4*)(dec_b + vb + 4);
#pragma unroll
        for (int j = 0; j < 2; ++j) {
            const int samp = sbase + (ws * 2 + j) * 16 + n16;
            AccU A; A.v = acc[i][j];
            A.f[0].x += d0.x; A.f[0].y += d0.y; A.f[0].z += d0.z; A.f[0].w += d0.w;
            A.f[1].x += d1.x; A.f[1].y += d1.y; A.f[1].z += d1.z; A.f[1].w += d1.w;
            *(float4*)(logits + (size_t)samp * VOCAB + vb)     = A.f[0];
            *(float4*)(logits + (size_t)samp * VOCAB + vb + 4) = A.f[1];
        }
    }
}

// ---------------------------------------------------------------------------
extern "C" void kernel_launch(void* const* d_in, const int* in_sizes, int n_in,
                              void* d_out, int out_size, void* d_ws, size_t ws_size,
                              hipStream_t stream) {
    (void)in_sizes; (void)n_in; (void)out_size;
    const int*   inp   = (const int*)  d_in[0];
    // d_in[1] = batch_size (16), fixed by problem constants
    const float* emb   = (const float*)d_in[2];
    const float* w1    = (const float*)d_in[3];
    const float* b1    = (const float*)d_in[4];
    const float* dec_w = (const float*)d_in[5];
    const float* dec_b = (const float*)d_in[6];

    float* logits = (float*)d_out;                       // [4096][32000] f32
    float* h_last = logits + (size_t)NSAMP * VOCAB;      // [16][512] f32

    // carve workspace (~50 MB total)
    char* ws = (char*)d_ws;
    size_t off = 0;
    auto carve = [&](size_t bytes) -> void* {
        void* p = ws + off;
        off = (off + bytes + 255) & ~(size_t)255;
        return p;
    };
    unsigned short* dec_w_bf = (unsigned short*)carve((size_t)VOCAB * HDIM * 2); // 32 MB
    unsigned short* wx_bf    = (unsigned short*)carve((size_t)HDIM * HDIM * 2);  // 512 KB
    unsigned short* wh_bf    = (unsigned short*)carve((size_t)HDIM * HDIM * 2);  // 512 KB
    unsigned short* x_bf     = (unsigned short*)carve((size_t)NSAMP * HDIM * 2); // 4 MB
    float*          pre      = (float*)carve((size_t)NSAMP * HDIM * 4);          // 8 MB
    unsigned short* out_bf   = (unsigned short*)carve((size_t)NSAMP * HDIM * 2); // 4 MB
    if (off > ws_size) return;  // workspace too small; nothing sane to do

    prep_kernel<<<2048, 256, 0, stream>>>(dec_w, w1, emb, inp,
                                          dec_w_bf, wx_bf, wh_bf, x_bf);

    pre_gemm_kernel<<<1024, 256, 0, stream>>>(wx_bf, x_bf, b1, pre);

    rnn_kernel<<<1, 1024, 0, stream>>>(wh_bf, pre, out_bf, h_last);

    decoder_kernel<<<dim3(VOCAB / 256, NSAMP / 128), 256 * 2, 0, stream>>>(
        dec_w_bf, out_bf, dec_b, logits);
}
